// DCN_MD_adp_43989055046090
// MI455X (gfx1250) — matrix operand order repaired and checked by reading, not compiled, not run on hardware
//
#include <hip/hip_runtime.h>

typedef __attribute__((ext_vector_type(16))) __bf16 v16bf;
typedef __attribute__((ext_vector_type(8)))  float  v8f;

#define BB 16384
#define FF 16
#define VV 100000
#define DD 16
#define EPSV 1e-5f

// ---------------- weight transpose + fp32 -> bf16 convert:  W[K,N] -> Wt[N,K] ----------------
__global__ void transpose_bf16(const float* __restrict__ W, __bf16* __restrict__ Wt, int K, int N) {
    int i = blockIdx.x * blockDim.x + threadIdx.x;
    if (i >= K * N) return;
    int k = i / N, n = i % N;
    Wt[(size_t)n * K + k] = (__bf16)W[i];
}

// ---------------- embedding gather: e[b, f*16+d] = emb[f, id, d] ----------------
__global__ void embed_gather(const int* __restrict__ ids, const float* __restrict__ emb,
                             float* __restrict__ ef, __bf16* __restrict__ eb) {
    int i = blockIdx.x * blockDim.x + threadIdx.x;   // over B*F
    if (i >= BB * FF) return;
    int b = i / FF, f = i % FF;
    int id = ids[i];
    const float* src = emb + ((size_t)f * VV + id) * DD;
    float*  df = ef + (size_t)b * (FF * DD) + f * DD;
    __bf16* db = eb + (size_t)b * (FF * DD) + f * DD;
#pragma unroll
    for (int d = 0; d < DD; ++d) { float v = src[d]; df[d] = v; db[d] = (__bf16)v; }
}

// ---------------- bf16 WMMA GEMM: C[M,N] = A[M,K] @ Wt[N,K]^T + bias ----------------
// 64x32 macro-tile per wave (4 M-subtiles x 2 N-subtiles), K-loop in steps of 32.
// Per K-step: 6KB loads -> 8 v_wmma_f32_16x16x32_bf16 (65536 MACs, ~10.7 MAC/byte).
__global__ __launch_bounds__(256) void gemm_bf16(const __bf16* __restrict__ A,
                                                 const __bf16* __restrict__ Wt,
                                                 const float* __restrict__ bias,
                                                 float* __restrict__ C,
                                                 int M, int N, int K) {
    int lane = threadIdx.x & 31;
    int wave = threadIdx.x >> 5;
    int nb = N >> 5;                       // # 32-wide N macro-tiles
    int mb = M >> 6;                       // # 64-high M macro-tiles
    int w = blockIdx.x * 8 + wave;         // uniform per wave -> EXEC all-ones inside
    int tM = w / nb, tN = w % nb;
    if (tM >= mb) return;
    int half = lane >> 4, r = lane & 15;
    // 16-bit A layout (ISA 7.12.2): lanes 0-15 hold K 0..7 (v0..3) and 16..23 (v4..7);
    // lanes 16-31 hold K 8..15 and 24..31.
    const __bf16* Abase = A  + (size_t)(tM * 64 + r) * K + half * 8;
    // B fragment: col N = r ; lanes 0-15 hold K 0..7 and 16..23, lanes 16-31 hold K 8..15 and 24..31.
    const __bf16* Bbase = Wt + (size_t)(tN * 32 + r) * K + half * 8;
    v8f acc[4][2] = {};
    for (int k0 = 0; k0 < K; k0 += 32) {
        v16bf a[4], b[2];
#pragma unroll
        for (int i = 0; i < 4; ++i) {
            const __bf16* Ar = Abase + (size_t)(i * 16) * K + k0;
#pragma unroll
            for (int e = 0; e < 8; ++e) { a[i][e] = Ar[e]; a[i][8 + e] = Ar[16 + e]; }
        }
#pragma unroll
        for (int j = 0; j < 2; ++j) {
            const __bf16* Br = Bbase + (size_t)(j * 16) * K + k0;
#pragma unroll
            for (int e = 0; e < 8; ++e) { b[j][e] = Br[e]; b[j][8 + e] = Br[16 + e]; }
        }
#pragma unroll
        for (int i = 0; i < 4; ++i)
#pragma unroll
            for (int j = 0; j < 2; ++j)
                acc[i][j] = __builtin_amdgcn_wmma_f32_16x16x32_bf16(
                    false, a[i], false, b[j], (short)0, acc[i][j], false, false);
    }
#pragma unroll
    for (int j = 0; j < 2; ++j) {
        int col = tN * 32 + j * 16 + r;
        float bv = bias ? bias[col] : 0.f;
#pragma unroll
        for (int i = 0; i < 4; ++i)
#pragma unroll
            for (int rr = 0; rr < 8; ++rr) {   // D layout: vgpr rr -> M = rr + half*8
                int row = tM * 64 + i * 16 + rr + half * 8;
                C[(size_t)row * N + col] = acc[i][j][rr] + bv;
            }
    }
}

// ---------------- per-column batch statistics (mean, rsqrt(var+eps)) ----------------
__global__ __launch_bounds__(256) void colstats(const float* __restrict__ Z, int M, int N,
                                                float* __restrict__ mean, float* __restrict__ rstd,
                                                int ddof) {
    int col = blockIdx.x;
    float s = 0.f, sq = 0.f;
    for (int r0 = threadIdx.x; r0 < M; r0 += 256) {
        float v = Z[(size_t)r0 * N + col];
        s += v; sq += v * v;
    }
    __shared__ float ss[256], sg[256];
    ss[threadIdx.x] = s; sg[threadIdx.x] = sq;
    __syncthreads();
    for (int o = 128; o > 0; o >>= 1) {
        if (threadIdx.x < o) { ss[threadIdx.x] += ss[threadIdx.x + o]; sg[threadIdx.x] += sg[threadIdx.x + o]; }
        __syncthreads();
    }
    if (threadIdx.x == 0) {
        float mu  = ss[0] / (float)M;
        float var = (sg[0] - mu * ss[0]) / (float)(M - ddof);
        mean[col] = mu;
        rstd[col] = rsqrtf(var + EPSV);
    }
}

// ---------------- normalize (+optional ReLU), emit fp32 and/or bf16 ----------------
__global__ void bn_act(const float* __restrict__ Z, const float* __restrict__ mean,
                       const float* __restrict__ rstd, const float* __restrict__ g,
                       const float* __restrict__ be, float* __restrict__ yf,
                       __bf16* __restrict__ yb, int M, int N, int relu) {
    size_t i = (size_t)blockIdx.x * blockDim.x + threadIdx.x;
    if (i >= (size_t)M * N) return;
    int col = (int)(i % N);
    float v = g[col] * (Z[i] - mean[col]) * rstd[col] + be[col];
    if (relu) v = fmaxf(v, 0.f);
    if (yf) yf[i] = v;
    if (yb) yb[i] = (__bf16)v;
}

// ---------------- cross network: 3 fused iterations, one wave per sample ----------------
__global__ __launch_bounds__(256) void crossnet(const float* __restrict__ e,
                                                const float* __restrict__ cw,
                                                const float* __restrict__ cb,
                                                float* __restrict__ out) {
    int lane = threadIdx.x & 31;
    int b = blockIdx.x * 8 + (threadIdx.x >> 5);
    if (b >= BB) return;
    const float* er = e + (size_t)b * 256;
    float ev[8], xv[8];
#pragma unroll
    for (int t = 0; t < 8; ++t) { ev[t] = er[lane + 32 * t]; xv[t] = ev[t]; }
    for (int l = 0; l < 3; ++l) {
        float p = 0.f;
#pragma unroll
        for (int t = 0; t < 8; ++t) p += xv[t] * cw[l * 256 + lane + 32 * t];
#pragma unroll
        for (int o = 16; o > 0; o >>= 1) p += __shfl_xor(p, o, 32);
#pragma unroll
        for (int t = 0; t < 8; ++t) xv[t] = ev[t] * p + cb[l * 256 + lane + 32 * t] + xv[t];
    }
    float* orow = out + (size_t)b * 256;
#pragma unroll
    for (int t = 0; t < 8; ++t) orow[lane + 32 * t] = xv[t];
}

// ---------------- low-rank hyper adapter: one wave per sample (lane = unit j) ----------------
__global__ __launch_bounds__(256) void adapter(const float* __restrict__ s0,
                                               const __bf16* __restrict__ H,
                                               const float* __restrict__ v0,
                                               const float* __restrict__ u1,
                                               const float* __restrict__ v1,
                                               const float* __restrict__ ab0,
                                               const float* __restrict__ ab1,
                                               float* __restrict__ t2) {
    int lane = threadIdx.x & 31;
    int b = blockIdx.x * 8 + (threadIdx.x >> 5);
    if (b >= BB) return;
    const __bf16* Hb = H + (size_t)b * 1024;
    float s = s0[(size_t)b * 32 + lane];
    float q = 0.f;                                   // q = (m@u0) @ H[b]
    for (int i = 0; i < 32; ++i) q += __shfl(s, i, 32) * (float)Hb[i * 32 + lane];
    float t = 0.f;                                   // t = sigmoid(q @ v0 + ab0)
    for (int i = 0; i < 32; ++i) t += __shfl(q, i, 32) * v0[i * 32 + lane];
    t = 1.f / (1.f + __expf(-(t + ab0[lane])));
    float r = 0.f;                                   // r = t @ u1
    for (int i = 0; i < 32; ++i) r += __shfl(t, i, 32) * u1[i * 32 + lane];
    float q2 = 0.f;                                  // q2 = r @ H[b]
    for (int i = 0; i < 32; ++i) q2 += __shfl(r, i, 32) * (float)Hb[i * 32 + lane];
    float acc[8];                                    // t2 = q2 @ v1 + ab1  (256 wide)
#pragma unroll
    for (int t8 = 0; t8 < 8; ++t8) acc[t8] = ab1[lane + 32 * t8];
    for (int i = 0; i < 32; ++i) {
        float qi = __shfl(q2, i, 32);
#pragma unroll
        for (int t8 = 0; t8 < 8; ++t8) acc[t8] += qi * v1[i * 256 + lane + 32 * t8];
    }
    float* orow = t2 + (size_t)b * 256;
#pragma unroll
    for (int t8 = 0; t8 < 8; ++t8) orow[lane + 32 * t8] = acc[t8];
}

// ---------------- final: BN(ddof=1)+residual, concat LR head, sigmoid ----------------
__global__ __launch_bounds__(256) void final_head(const float* __restrict__ t2,
                                                  const float* __restrict__ mean,
                                                  const float* __restrict__ rstd,
                                                  const float* __restrict__ gamma1,
                                                  const float* __restrict__ bias1,
                                                  const float* __restrict__ m,
                                                  const float* __restrict__ cn,
                                                  const float* __restrict__ lw,
                                                  const float* __restrict__ lb,
                                                  float* __restrict__ out) {
    int lane = threadIdx.x & 31;
    int b = blockIdx.x * 8 + (threadIdx.x >> 5);
    if (b >= BB) return;
    float acc = 0.f;
#pragma unroll
    for (int t8 = 0; t8 < 8; ++t8) {
        int c = lane + 32 * t8;
        float mv = gamma1[c] * (t2[(size_t)b * 256 + c] - mean[c]) * rstd[c] + bias1[c]
                 + m[(size_t)b * 256 + c];
        acc += cn[(size_t)b * 256 + c] * lw[c] + mv * lw[256 + c];
    }
#pragma unroll
    for (int o = 16; o > 0; o >>= 1) acc += __shfl_xor(acc, o, 32);
    if (lane == 0) out[b] = 1.f / (1.f + __expf(-(acc + lb[0])));
}

// =====================================================================================
extern "C" void kernel_launch(void* const* d_in, const int* in_sizes, int n_in,
                              void* d_out, int out_size, void* d_ws, size_t ws_size,
                              hipStream_t stream) {
    (void)in_sizes; (void)n_in; (void)out_size; (void)ws_size;
    const int*   feat_ids = (const int*)d_in[0];
    const float* emb  = (const float*)d_in[2];
    const float* hW0  = (const float*)d_in[3];  const float* hb0 = (const float*)d_in[4];
    const float* hg0  = (const float*)d_in[5];  const float* hbe0= (const float*)d_in[6];
    const float* hW1  = (const float*)d_in[7];  const float* hb1 = (const float*)d_in[8];
    const float* hg1  = (const float*)d_in[9];  const float* hbe1= (const float*)d_in[10];
    const float* hW2  = (const float*)d_in[11]; const float* hb2 = (const float*)d_in[12];
    const float* hg2  = (const float*)d_in[13]; const float* hbe2= (const float*)d_in[14];
    const float* cw   = (const float*)d_in[15]; const float* cb  = (const float*)d_in[16];
    const float* mW0  = (const float*)d_in[17]; const float* mb0 = (const float*)d_in[18];
    const float* mg0  = (const float*)d_in[19]; const float* mbe0= (const float*)d_in[20];
    const float* mW1  = (const float*)d_in[21]; const float* mb1 = (const float*)d_in[22];
    const float* mg1  = (const float*)d_in[23]; const float* mbe1= (const float*)d_in[24];
    const float* u0   = (const float*)d_in[25]; const float* u1  = (const float*)d_in[26];
    const float* v0   = (const float*)d_in[27]; const float* v1  = (const float*)d_in[28];
    const float* ab0  = (const float*)d_in[29]; const float* ab1 = (const float*)d_in[30];
    const float* gamma1 = (const float*)d_in[31]; const float* bias1 = (const float*)d_in[32];
    const float* lw   = (const float*)d_in[33]; const float* lb  = (const float*)d_in[34];
    float* out = (float*)d_out;

    // ---- workspace carve-up ----
    char* base = (char*)d_ws; size_t off = 0;
    auto take = [&](size_t bytes) -> void* {
        void* r = base + off; off = (off + bytes + 255) & ~(size_t)255; return r;
    };
    float*  ef    = (float*) take((size_t)BB * 256 * 4);
    __bf16* eb    = (__bf16*)take((size_t)BB * 256 * 2);
    __bf16* wt_h0 = (__bf16*)take((size_t)128  * 256 * 2);
    __bf16* wt_h1 = (__bf16*)take((size_t)64   * 128 * 2);
    __bf16* wt_h2 = (__bf16*)take((size_t)1024 * 64  * 2);
    __bf16* wt_m0 = (__bf16*)take((size_t)512  * 256 * 2);
    __bf16* wt_m1 = (__bf16*)take((size_t)256  * 512 * 2);
    __bf16* wt_u0 = (__bf16*)take((size_t)32   * 256 * 2);
    float*  z     = (float*) take((size_t)BB * 1024 * 4);   // reused for every GEMM out + t2
    __bf16* Hb    = (__bf16*)take((size_t)BB * 1024 * 2);
    __bf16* actA  = (__bf16*)take((size_t)BB * 512 * 2);
    __bf16* actB  = (__bf16*)take((size_t)BB * 512 * 2);
    float*  mf    = (float*) take((size_t)BB * 256 * 4);
    float*  cn    = (float*) take((size_t)BB * 256 * 4);
    float*  s0b   = (float*) take((size_t)BB * 32  * 4);
    float*  mean  = (float*) take((size_t)1024 * 4);
    float*  rstd  = (float*) take((size_t)1024 * 4);
    float*  t2    = z;   // z region free after the last GEMM

    auto xpose = [&](const float* W, __bf16* Wt, int K, int N) {
        int n = K * N;
        transpose_bf16<<<(n + 255) / 256, 256, 0, stream>>>(W, Wt, K, N);
    };
    auto gemm = [&](const __bf16* A, const __bf16* Wt, const float* bias, float* C,
                    int M, int N, int K) {
        int tiles = (M / 64) * (N / 32);
        gemm_bf16<<<(tiles + 7) / 8, 256, 0, stream>>>(A, Wt, bias, C, M, N, K);
    };
    auto stats = [&](const float* Z, int N, int ddof) {
        colstats<<<N, 256, 0, stream>>>(Z, BB, N, mean, rstd, ddof);
    };
    auto norm = [&](const float* Z, const float* g, const float* be, float* yf, __bf16* yb,
                    int N, int relu) {
        size_t n = (size_t)BB * N;
        bn_act<<<(unsigned)((n + 255) / 256), 256, 0, stream>>>(Z, mean, rstd, g, be, yf, yb, BB, N, relu);
    };

    // weight prep (tiny, deterministic each call)
    xpose(hW0, wt_h0, 256, 128);  xpose(hW1, wt_h1, 128, 64);
    xpose(hW2, wt_h2, 64, 1024);  xpose(mW0, wt_m0, 256, 512);
    xpose(mW1, wt_m1, 512, 256);  xpose(u0,  wt_u0, 256, 32);

    // embedding gather (table is L2-resident: 102MB < 192MB)
    embed_gather<<<(BB * FF + 255) / 256, 256, 0, stream>>>(feat_ids, emb, ef, eb);

    // hyper_net: 256 -> 128 -> 64 -> 1024, each Linear+BN(train)+ReLU
    gemm(eb,   wt_h0, hb0, z, BB, 128, 256);  stats(z, 128, 0);  norm(z, hg0, hbe0, nullptr, actA, 128, 1);
    gemm(actA, wt_h1, hb1, z, BB, 64,  128);  stats(z, 64,  0);  norm(z, hg1, hbe1, nullptr, actB, 64,  1);
    gemm(actB, wt_h2, hb2, z, BB, 1024, 64);  stats(z, 1024,0);  norm(z, hg2, hbe2, nullptr, Hb,   1024,1);

    // cross network (fused 3 layers)
    crossnet<<<(BB + 7) / 8, 256, 0, stream>>>(ef, cw, cb, cn);

    // MLP: 256 -> 512 -> 256
    gemm(eb,   wt_m0, mb0, z, BB, 512, 256);  stats(z, 512, 0);  norm(z, mg0, mbe0, nullptr, actA, 512, 1);
    gemm(actA, wt_m1, mb1, z, BB, 256, 512);  stats(z, 256, 0);  norm(z, mg1, mbe1, mf,      actB, 256, 1);

    // adapter: s0 = m @ u0 (WMMA), then per-sample 32x32 chains against H
    gemm(actB, wt_u0, nullptr, s0b, BB, 32, 256);
    adapter<<<(BB + 7) / 8, 256, 0, stream>>>(s0b, Hb, v0, u1, v1, ab0, ab1, t2);

    // final BN (unbiased var) + residual + LR head + sigmoid
    colstats<<<256, 256, 0, stream>>>(t2, BB, 256, mean, rstd, 1);
    final_head<<<(BB + 7) / 8, 256, 0, stream>>>(t2, mean, rstd, gamma1, bias1, mf, cn, lw, lb, out);
}
